// PointCloud3DConv_8967891714396
// MI455X (gfx1250) — compile-verified, matrix operand order checked by reading, not run on hardware
//
#include <hip/hip_runtime.h>
#include <hip/hip_bf16.h>

#define BB   2
#define NPTS 150000
#define BN   (BB * NPTS)
#define CC   64
#define SX   128
#define EPSV 1e-5f

typedef __attribute__((ext_vector_type(16))) __bf16   v16bf;
typedef __attribute__((ext_vector_type(8)))  float    v8f;
typedef __attribute__((ext_vector_type(4)))  uint32_t v4u;
typedef __attribute__((ext_vector_type(8)))  uint32_t v8u;

// ---------------- workspace layout (bytes) ----------------
static const size_t SZ_GRID  = (size_t)BB * SX * SX * SX * 4;      // 16,777,216
static const size_t OFF_FEAT = SZ_GRID;
static const size_t SZ_FEAT  = (size_t)(BN + 16) * CC * 2;         // bf16 rows + zero pad
static const size_t OFF_W    = OFF_FEAT + SZ_FEAT;
static const size_t SZ_W     = 27ull * CC * CC * 2;                // bf16 [k][cout][cin]
static const size_t OFF_PART = OFF_W + SZ_W;
static const int    NPBLK    = (BN + 511) / 512;                   // 586
static const size_t SZ_PART  = (size_t)NPBLK * 128 * 4;
static const size_t OFF_STAT = OFF_PART + SZ_PART;                 // 128 floats

// ---------------- small utility kernels ----------------
__global__ void k_grid_init(int* __restrict__ g) {
  size_t total = (size_t)BB * SX * SX * SX;
  for (size_t i = (size_t)blockIdx.x * blockDim.x + threadIdx.x; i < total;
       i += (size_t)gridDim.x * blockDim.x)
    g[i] = -1;
}

__global__ void k_scatter(const int* __restrict__ coords, int* __restrict__ g) {
  int p = blockIdx.x * blockDim.x + threadIdx.x;
  if (p >= BN) return;
  int b = p / NPTS;
  int x = coords[p * 3 + 0], y = coords[p * 3 + 1], z = coords[p * 3 + 2];
  atomicMax(&g[((b * SX + x) * SX + y) * SX + z], p);
}

__global__ void k_cvt_feat(const float* __restrict__ f, __bf16* __restrict__ fb) {
  size_t total = (size_t)(BN + 16) * CC;
  size_t lim   = (size_t)BN * CC;
  for (size_t i = (size_t)blockIdx.x * blockDim.x + threadIdx.x; i < total;
       i += (size_t)gridDim.x * blockDim.x)
    fb[i] = (i < lim) ? (__bf16)f[i] : (__bf16)0.0f;
}

// weight: [27][cin][cout] f32  ->  [27][cout][cin] bf16 (transposed for B-frags)
__global__ void k_cvt_w(const float* __restrict__ w, __bf16* __restrict__ wt) {
  int i = blockIdx.x * blockDim.x + threadIdx.x;
  if (i >= 27 * CC * CC) return;
  int k  = i >> 12;
  int co = (i >> 6) & 63;
  int ci = i & 63;
  wt[i] = (__bf16)w[(k << 12) + (ci << 6) + co];
}

// ---------------- WMMA fragment helper ----------------
// 16-bit A/B layout (ISA 7.12.2): lanes 0-15 hold K = {kb+0..7, kb+16..23},
// lanes 16-31 hold K = {kb+8..15, kb+24..31}.
__device__ __forceinline__ v16bf load_frag16(const __bf16* rowptr, int kbase, int laneHalf) {
  v16bf v;
  uint4* vp = (uint4*)&v;
  vp[0] = *(const uint4*)(rowptr + kbase + (laneHalf << 3));
  vp[1] = *(const uint4*)(rowptr + kbase + 16 + (laneHalf << 3));
  return v;
}

// ---------------- async staging of one tap ----------------
// A tile: per-thread async gather of 64B straight into LDS (ASYNCcnt).
// W tile: one TDM descriptor per tap, hardware pads 128B rows by 16B -> stride 144B.
#define LDA 72
__device__ __forceinline__ void stage_tap(
    int kk, uint32_t ldsA, uint32_t ldsW,
    const __bf16* __restrict__ feats, const __bf16* __restrict__ wts,
    const int* __restrict__ grid, unsigned vmask, int gbase, int gh, bool leader)
{
  // neighbor id for this thread's tile row
  int id = BN;  // zero row
  if ((vmask >> kk) & 1u) {
    int ox = kk / 9 - 1, oy = (kk / 3) % 3 - 1, oz = kk % 3 - 1;
    int g = grid[gbase + ox * (SX * SX) + oy * SX + oz];
    if (g >= 0) id = g;
  }
  unsigned long long ga =
      (unsigned long long)(uintptr_t)(feats + (size_t)id * CC + gh * 32);
  asm volatile(
      "global_load_async_to_lds_b128 %0, %1, off\n\t"
      "global_load_async_to_lds_b128 %0, %1, off offset:16\n\t"
      "global_load_async_to_lds_b128 %0, %1, off offset:32\n\t"
      "global_load_async_to_lds_b128 %0, %1, off offset:48"
      :: "v"(ldsA), "v"(ga) : "memory");

  if (leader) {
    unsigned long long wa =
        (unsigned long long)(uintptr_t)(wts + (size_t)kk * CC * CC);
    v4u g0; v8u g1; v4u gz = (v4u){0u, 0u, 0u, 0u};
    g0[0] = 1u;                                   // count=1 valid descriptor
    g0[1] = ldsW;                                 // lds_addr
    g0[2] = (uint32_t)wa;                         // global_addr[31:0]
    g0[3] = (uint32_t)((wa >> 32) & 0x01FFFFFFu)  // global_addr[56:32]
            | (2u << 30);                         // type=2 (image)
    g1[0] = (1u << 16)      // data_size = 2 bytes
          | (1u << 20)      // pad_enable
          | (4u << 22)      // pad_interval: 32 DWORDs (one 128B row)
          | (3u << 25);     // pad_amount: 4 DWORDs (16B) -> LDS stride 144B
    g1[1] = 64u << 16;      // tensor_dim0 = 64
    g1[2] = 64u << 16;      // tensor_dim1 = 64
    g1[3] = 64u << 16;      // tile_dim0 = 64
    g1[4] = 64u;            // tile_dim1 = 64, tile_dim2 = 0
    g1[5] = 64u;            // tensor_dim0_stride = 64
    g1[6] = 0u;
    g1[7] = 0u;
    asm volatile("tensor_load_to_lds %0, %1, %2, %3"
                 :: "s"(g0), "s"(g1), "s"(gz), "s"(gz) : "memory");
  }
}

// ---------------- fused rulebook-gather + GEMM ----------------
__global__ __launch_bounds__(256) void k_gemm(
    const __bf16* __restrict__ feats,   // (BN+16) x 64 bf16, row BN == zeros
    const __bf16* __restrict__ wts,     // 27 x 64(cout) x 64(cin) bf16
    const int*    __restrict__ grid,    // B x 128^3, -1 or max pid
    const int*    __restrict__ coords,  // BN x 3
    float*        __restrict__ out)     // BN x 64 raw conv output
{
  __shared__ __bf16 lA[2][128 * LDA];
  __shared__ __bf16 lW[2][64 * LDA];

  const int t        = threadIdx.x;
  const int wave     = t >> 5;
  const int lane     = t & 31;
  const int laneHalf = lane >> 4;
  const int laneMod  = lane & 15;
  const int row0     = blockIdx.x * 128;

  // gather role: 2 threads per tile row
  const int gr = t >> 1;
  const int gh = t & 1;
  const int p  = row0 + gr;

  int gbase = 0;
  unsigned vmask = 0;
  if (p < BN) {
    int cb = p / NPTS;
    int cx = coords[p * 3 + 0];
    int cy = coords[p * 3 + 1];
    int cz = coords[p * 3 + 2];
    gbase = ((cb * SX + cx) * SX + cy) * SX + cz;
#pragma unroll
    for (int kk = 0; kk < 27; ++kk) {
      int nx = cx + (kk / 9 - 1), ny = cy + ((kk / 3) % 3 - 1), nz = cz + (kk % 3 - 1);
      if ((unsigned)nx < (unsigned)SX && (unsigned)ny < (unsigned)SX &&
          (unsigned)nz < (unsigned)SX)
        vmask |= 1u << kk;
    }
  }

  const uint32_t ldsA[2] = {
      (uint32_t)(uintptr_t)&lA[0][gr * LDA + gh * 32],
      (uint32_t)(uintptr_t)&lA[1][gr * LDA + gh * 32]};
  const uint32_t ldsW[2] = {(uint32_t)(uintptr_t)&lW[0][0],
                            (uint32_t)(uintptr_t)&lW[1][0]};
  const bool leader = (__builtin_amdgcn_readfirstlane(wave) == 0);

  v8f acc[4];
#pragma unroll
  for (int n = 0; n < 4; ++n) acc[n] = (v8f){0.f, 0.f, 0.f, 0.f, 0.f, 0.f, 0.f, 0.f};

  // prologue: stage tap 0 into buffer 0
  stage_tap(0, ldsA[0], ldsW[0], feats, wts, grid, vmask, gbase, gh, leader);
  asm volatile("s_wait_asynccnt 0x0" ::: "memory");
  if (leader) __builtin_amdgcn_s_wait_tensorcnt(0);
  __syncthreads();

  for (int kk = 0; kk < 27; ++kk) {
    const int buf = kk & 1;
    // overlap: stage next tap into the other buffer while computing this one
    if (kk + 1 < 27)
      stage_tap(kk + 1, ldsA[buf ^ 1], ldsW[buf ^ 1], feats, wts, grid, vmask,
                gbase, gh, leader);

    const __bf16* arow = &lA[buf][(wave * 16 + laneMod) * LDA];
    v16bf a0 = load_frag16(arow, 0, laneHalf);
    v16bf a1 = load_frag16(arow, 32, laneHalf);
#pragma unroll
    for (int n = 0; n < 4; ++n) {
      const __bf16* bcol = &lW[buf][(n * 16 + laneMod) * LDA];
      v16bf b0 = load_frag16(bcol, 0, laneHalf);
      v16bf b1 = load_frag16(bcol, 32, laneHalf);
      acc[n] = __builtin_amdgcn_wmma_f32_16x16x32_bf16(
          false, a0, false, b0, (short)0, acc[n], false, false);
      acc[n] = __builtin_amdgcn_wmma_f32_16x16x32_bf16(
          false, a1, false, b1, (short)0, acc[n], false, false);
    }

    // publish next buffer: my async gathers + leader's TDM, then one barrier
    asm volatile("s_wait_asynccnt 0x0" ::: "memory");
    if (leader) __builtin_amdgcn_s_wait_tensorcnt(0);
    __syncthreads();
  }

  // store raw conv output (C/D layout: VGPR r -> M = r + 8*laneHalf)
#pragma unroll
  for (int n = 0; n < 4; ++n) {
#pragma unroll
    for (int r = 0; r < 8; ++r) {
      int row = row0 + wave * 16 + laneHalf * 8 + r;
      if (row < BN) out[(size_t)row * CC + n * 16 + laneMod] = acc[n][r];
    }
  }
}

// ---------------- batch-norm statistics (deterministic 2-pass) ----------------
__global__ __launch_bounds__(256) void k_stats_part(const float* __restrict__ out,
                                                    float* __restrict__ part) {
  __shared__ float s1[256], s2[256];
  int t = threadIdx.x;
  int c = t & 63;
  int g = t >> 6;
  size_t rbase = (size_t)blockIdx.x * 512;
  float s = 0.f, q = 0.f;
  for (int r = g; r < 512; r += 4) {
    size_t row = rbase + r;
    if (row < BN) {
      float v = out[row * CC + c];
      s += v;
      q += v * v;
    }
  }
  s1[t] = s; s2[t] = q;
  __syncthreads();
  if (t < 64) {
    float ts = s1[t] + s1[t + 64] + s1[t + 128] + s1[t + 192];
    float tq = s2[t] + s2[t + 64] + s2[t + 128] + s2[t + 192];
    part[(size_t)blockIdx.x * 128 + t]      = ts;
    part[(size_t)blockIdx.x * 128 + 64 + t] = tq;
  }
}

__global__ void k_stats_final(const float* __restrict__ part,
                              const float* __restrict__ gamma,
                              float* __restrict__ stat, int nblk) {
  int c = threadIdx.x;
  if (c >= 64) return;
  float s = 0.f, q = 0.f;
  for (int b = 0; b < nblk; ++b) {
    s += part[(size_t)b * 128 + c];
    q += part[(size_t)b * 128 + 64 + c];
  }
  float mean = s / (float)BN;
  float var  = q / (float)BN - mean * mean;
  float inv  = rsqrtf(var + EPSV);
  stat[c]      = mean;
  stat[64 + c] = inv * gamma[c];
}

__global__ void k_norm_relu(float* __restrict__ out, const float* __restrict__ stat,
                            const float* __restrict__ beta) {
  size_t total = (size_t)BN * CC;
  for (size_t i = (size_t)blockIdx.x * blockDim.x + threadIdx.x; i < total;
       i += (size_t)gridDim.x * blockDim.x) {
    int c = (int)(i & 63);
    float v = (out[i] - stat[c]) * stat[64 + c] + beta[c];
    out[i] = v > 0.f ? v : 0.f;
  }
}

// ---------------- launcher ----------------
extern "C" void kernel_launch(void* const* d_in, const int* in_sizes, int n_in,
                              void* d_out, int out_size, void* d_ws, size_t ws_size,
                              hipStream_t stream) {
  const float* features = (const float*)d_in[0];
  const int*   coords   = (const int*)d_in[1];
  const float* weight   = (const float*)d_in[2];
  const float* gamma    = (const float*)d_in[3];
  const float* beta     = (const float*)d_in[4];
  float*       out      = (float*)d_out;

  char* ws = (char*)d_ws;
  int*    grid    = (int*)(ws);
  __bf16* featsBf = (__bf16*)(ws + OFF_FEAT);
  __bf16* wBf     = (__bf16*)(ws + OFF_W);
  float*  part    = (float*)(ws + OFF_PART);
  float*  stat    = (float*)(ws + OFF_STAT);

  k_grid_init<<<4096, 256, 0, stream>>>(grid);
  k_scatter<<<(BN + 255) / 256, 256, 0, stream>>>(coords, grid);

  k_cvt_feat<<<8192, 256, 0, stream>>>(features, featsBf);
  k_cvt_w<<<(27 * CC * CC + 255) / 256, 256, 0, stream>>>(weight, wBf);

  k_gemm<<<(BN + 127) / 128, 256, 0, stream>>>(featsBf, wBf, grid, coords, out);

  k_stats_part<<<NPBLK, 256, 0, stream>>>(out, part);
  k_stats_final<<<1, 64, 0, stream>>>(part, gamma, stat, NPBLK);
  k_norm_relu<<<8192, 256, 0, stream>>>(out, stat, beta);
}
